// MACELayer_66460323938668
// MI455X (gfx1250) — compile-verified, matrix operand order checked by reading, not run on hardware
//
#include <hip/hip_runtime.h>
#include <hip/hip_bf16.h>

typedef __attribute__((ext_vector_type(16))) _Float16 v16h;
typedef __attribute__((ext_vector_type(8)))  _Float16 v8h;
typedef __attribute__((ext_vector_type(8)))  float    v8f;

#define ROW   256           // interleaved row: [s0(64) | vec (3*c+m) (192)]
#define INV3  0.57735026918962576451f
#define INV2  0.70710678118654752440f
#define NW    4             // waves per block (wave32 -> 128 threads)

// --------- packed weight-fragment buffer layout (units: fragments of 512 halves = 1KB) ---------
#define FR_PRE_W0 0     // 8  : j*2+kk
#define FR_PRE_W1 8     // 8
#define FR_MLP_W1 16    // 4  : j
#define FR_MLP_W2 20    // 8  : j*2+kk
#define FR_MLP_W3 28    // 40 : t*2+kk, t = col16 tile 0..19
#define FR_LIN0   68    // 8
#define FR_LIN1   76    // 8
#define FR_C0W    84    // 12 : t (0..11)
#define FR_C1W    96    // 8  : t (0..7)
#define FR_SCW0   104   // 80 : ks*4+j
#define FR_SCW1   184   // 80
#define FR_FIN0   264   // 8
#define FR_FIN1   272   // 8
#define FR_TOTAL  280

// ---------------- WMMA helpers ----------------

__device__ __forceinline__ v8f wmma_f16(v16h a, v16h b, v8f c) {
    return __builtin_amdgcn_wmma_f32_16x16x32_f16(
        false, a, false, b, (short)0, c, false, false);
}

// K index inside a 32-wide fragment for a given lane/half slot
__device__ __forceinline__ int kmap(int lane, int h) {
    return ((lane >> 4) << 3) + (h < 8 ? h : h + 8);
}

// A fragment from row-major f32 [rows x stride]; tile assumed full; K zero-padded.
__device__ __forceinline__ v16h fragA_rows(const float* __restrict__ base, int stride,
                                           int row0, int k0, int kmax, int lane) {
    const float* p = base + (long)(row0 + (lane & 15)) * stride;
    v16h a;
#pragma unroll
    for (int h = 0; h < 16; ++h) {
        int k = k0 + kmap(lane, h);
        a[h] = (_Float16)((k < kmax) ? p[k] : 0.0f);
    }
    return a;
}

// A fragment of vec component m from interleaved (*,256) buffer: elem = p[64+3k+m]
__device__ __forceinline__ v16h fragA_vec3(const float* __restrict__ base,
                                           int row0, int m, int k0, int lane) {
    const float* p = base + (long)(row0 + (lane & 15)) * ROW + 64 + m;
    v16h a;
#pragma unroll
    for (int h = 0; h < 16; ++h) {
        int k = k0 + kmap(lane, h);
        a[h] = (_Float16)p[3 * k];
    }
    return a;
}

// B fragment from row-major f32 [K x stride]; used only by the pack kernel.
__device__ __forceinline__ v16h fragB(const float* __restrict__ base, int stride,
                                      int k0, int col0, int kmax, int lane) {
    int col = col0 + (lane & 15);
    v16h b;
#pragma unroll
    for (int h = 0; h < 16; ++h) {
        int k = k0 + kmap(lane, h);
        b[h] = (_Float16)((k < kmax) ? base[(long)k * stride + col] : 0.0f);
    }
    return b;
}

// B fragment from sc_W (C,A,C): flat K = a*64 + c maps to W[(c*10+a)*64 + col]
__device__ __forceinline__ v16h fragB_scw(const float* __restrict__ W,
                                          int a, int c0, int col0, int lane) {
    int col = col0 + (lane & 15);
    v16h b;
#pragma unroll
    for (int h = 0; h < 16; ++h) {
        int c = c0 + kmap(lane, h);
        b[h] = (_Float16)W[((long)c * 10 + a) * 64 + col];
    }
    return b;
}

// Packed B fragment: 32 lanes x 32B contiguous -> two b128 loads
__device__ __forceinline__ v16h fragB_pk(const _Float16* __restrict__ fb, int fragIdx, int lane) {
    return *((const v16h*)(fb + (long)fragIdx * 512) + lane);
}

// A fragment from f16 row-major LDS tile [16 x stride]; stride*2 must be 16B aligned.
// Per lane: halves 0..7 = K kbase..kbase+7, halves 8..15 = K kbase+16..kbase+23
// -> exactly two contiguous 16B runs -> two ds_load_b128, no converts.
__device__ __forceinline__ v16h fragA_lds16row(const _Float16* buf, int stride, int k0, int lane) {
    const _Float16* p = buf + (lane & 15) * stride + k0 + ((lane >> 4) << 3);
    v8h lo = *(const v8h*)p;
    v8h hi = *(const v8h*)(p + 16);
    return __builtin_shufflevector(lo, hi, 0, 1, 2, 3, 4, 5, 6, 7,
                                          8, 9, 10, 11, 12, 13, 14, 15);
}

// A fragment from f16 LDS tile, interleaved vec part: elem = p[64+3k+m] (scalar path)
__device__ __forceinline__ v16h fragA_lds16v(const _Float16* buf, int stride, int k0,
                                             int lane, int m) {
    const _Float16* p = buf + (lane & 15) * stride + 64 + m;
    v16h a;
#pragma unroll
    for (int h = 0; h < 16; ++h) {
        int k = k0 + kmap(lane, h);
        a[h] = p[3 * k];
    }
    return a;
}

// packed f16 scale: frag * scalar (lowers to v_pk_mul_f16)
__device__ __forceinline__ v16h vsplat_mul(v16h x, _Float16 s) {
    v16h r;
#pragma unroll
    for (int h = 0; h < 16; ++h) r[h] = x[h] * s;
    return r;
}

__device__ __forceinline__ v8f silu8(v8f x) {
    v8f r;
#pragma unroll
    for (int i = 0; i < 8; ++i) {
        float v = x[i];
        r[i] = v / (1.0f + __expf(-v));
    }
    return r;
}

// ---------------- kernels ----------------

__global__ void zero_kernel(float4* __restrict__ p, long n4) {
    long i = (long)blockIdx.x * blockDim.x + threadIdx.x;
    if (i < n4) p[i] = make_float4(0.f, 0.f, 0.f, 0.f);
}

// Pre-pack all weight matrices into f16 B-fragments (one fragment per block).
__global__ void pack_kernel(const float* __restrict__ preW0, const float* __restrict__ preW1,
                            const float* __restrict__ W1,    const float* __restrict__ W2,
                            const float* __restrict__ W3,    const float* __restrict__ lin0,
                            const float* __restrict__ lin1,  const float* __restrict__ c0w,
                            const float* __restrict__ c1w,   const float* __restrict__ scW0,
                            const float* __restrict__ scW1,  const float* __restrict__ fin0,
                            const float* __restrict__ fin1,  _Float16* __restrict__ fb) {
    int f = blockIdx.x;
    int lane = threadIdx.x & 31;
    v16h v;
    if      (f < FR_PRE_W1) { int g = f - FR_PRE_W0; v = fragB(preW0, 64, (g & 1) * 32, (g >> 1) * 16, 64, lane); }
    else if (f < FR_MLP_W1) { int g = f - FR_PRE_W1; v = fragB(preW1, 64, (g & 1) * 32, (g >> 1) * 16, 64, lane); }
    else if (f < FR_MLP_W2) { int g = f - FR_MLP_W1; v = fragB(W1,    64, 0,            g * 16,        8,  lane); }
    else if (f < FR_MLP_W3) { int g = f - FR_MLP_W2; v = fragB(W2,    64, (g & 1) * 32, (g >> 1) * 16, 64, lane); }
    else if (f < FR_LIN0)   { int g = f - FR_MLP_W3; v = fragB(W3,   320, (g & 1) * 32, (g >> 1) * 16, 64, lane); }
    else if (f < FR_LIN1)   { int g = f - FR_LIN0;   v = fragB(lin0,  64, (g & 1) * 32, (g >> 1) * 16, 64, lane); }
    else if (f < FR_C0W)    { int g = f - FR_LIN1;   v = fragB(lin1,  64, (g & 1) * 32, (g >> 1) * 16, 64, lane); }
    else if (f < FR_C1W)    { int g = f - FR_C0W;    v = fragB(c0w,  192, 0,            g * 16,        10, lane); }
    else if (f < FR_SCW0)   { int g = f - FR_C1W;    v = fragB(c1w,  128, 0,            g * 16,        10, lane); }
    else if (f < FR_SCW1)   { int g = f - FR_SCW0;   int ks = g >> 2, j = g & 3;
                              v = fragB_scw(scW0, ks >> 1, (ks & 1) * 32, j * 16, lane); }
    else if (f < FR_FIN0)   { int g = f - FR_SCW1;   int ks = g >> 2, j = g & 3;
                              v = fragB_scw(scW1, ks >> 1, (ks & 1) * 32, j * 16, lane); }
    else if (f < FR_FIN1)   { int g = f - FR_FIN0;   v = fragB(fin0,  64, (g & 1) * 32, (g >> 1) * 16, 64, lane); }
    else                    { int g = f - FR_FIN1;   v = fragB(fin1,  64, (g & 1) * 32, (g >> 1) * 16, 64, lane); }
    *((v16h*)(fb + (long)f * 512) + lane) = v;
}

// H[n, 0:64] = x0 @ pre_W0 ; H[n, 64+3d+m] = sum_c x1[n,c,m] pre_W1[c,d]
__global__ void pre_kernel(const float* __restrict__ nf, const _Float16* __restrict__ fb,
                           float* __restrict__ H, int N_) {
    int lane = threadIdx.x & 31;
    int wv   = threadIdx.x >> 5;
    int nBase = (blockIdx.x * NW + wv) * 16;
    if (nBase + 16 > N_) return;                  // wave-uniform exit (16 | N)
    int ch = lane & 15;
    int hi = (lane >> 4) << 3;
    float* h0p = H + (long)(nBase + hi) * ROW + ch;           // + r*ROW + j*16
    float* h1p = H + (long)(nBase + hi) * ROW + 64 + 3 * ch;  // + r*ROW + 48*j + m

    v16h a0a = fragA_rows(nf, ROW, nBase, 0,  64, lane);
    v16h a0b = fragA_rows(nf, ROW, nBase, 32, 64, lane);
#pragma unroll
    for (int j = 0; j < 4; ++j) {
        v8f z = {};
        z = wmma_f16(a0a, fragB_pk(fb, FR_PRE_W0 + j * 2 + 0, lane), z);
        z = wmma_f16(a0b, fragB_pk(fb, FR_PRE_W0 + j * 2 + 1, lane), z);
#pragma unroll
        for (int r = 0; r < 8; ++r)
            h0p[r * ROW + j * 16] = z[r];
    }
#pragma unroll
    for (int m = 0; m < 3; ++m) {
        v16h a1a = fragA_vec3(nf, nBase, m, 0,  lane);
        v16h a1b = fragA_vec3(nf, nBase, m, 32, lane);
#pragma unroll
        for (int j = 0; j < 4; ++j) {
            v8f z = {};
            z = wmma_f16(a1a, fragB_pk(fb, FR_PRE_W1 + j * 2 + 0, lane), z);
            z = wmma_f16(a1b, fragB_pk(fb, FR_PRE_W1 + j * 2 + 1, lane), z);
#pragma unroll
            for (int r = 0; r < 8; ++r)
                h1p[r * ROW + 48 * j + m] = z[r];
        }
    }
}

// Fused: radial MLP (WMMA) -> messages -> atomic scatter into T(N,256)
__global__ void edge_kernel(const float* __restrict__ H, const float* __restrict__ sph,
                            const float* __restrict__ rb, const int* __restrict__ ei,
                            const _Float16* __restrict__ fb, float* __restrict__ T,
                            int E_) {
    __shared__ __align__(16) _Float16 sHid[NW][16][72];   // f16 row-major, 144B stride
    __shared__ int   sSnd[NW][16];
    __shared__ int   sRcv[NW][16];
    __shared__ float sY[NW][16][4];

    int lane = threadIdx.x & 31;
    int wv   = threadIdx.x >> 5;
    int eBase = (blockIdx.x * NW + wv) * 16;
    if (eBase + 16 > E_) return;                  // terminated waves don't block barriers
    int ch = lane & 15;
    int hi = (lane >> 4) << 3;

    if (lane < 16) {
        int e = eBase + lane;
        sSnd[wv][lane] = ei[e];
        sRcv[wv][lane] = ei[E_ + e];
        sY[wv][lane][0] = sph[(long)e * 4 + 0];
        sY[wv][lane][1] = sph[(long)e * 4 + 1];
        sY[wv][lane][2] = sph[(long)e * 4 + 2];
        sY[wv][lane][3] = sph[(long)e * 4 + 3];
    }
    __syncthreads();

    // layer 1: (16x8) @ (8x64), K padded to 32
    v16h a1 = fragA_rows(rb, 8, eBase, 0, 8, lane);
    v8f acc[4];
#pragma unroll
    for (int j = 0; j < 4; ++j) {
        v8f z = {};
        z = wmma_f16(a1, fragB_pk(fb, FR_MLP_W1 + j, lane), z);
        acc[j] = silu8(z);
    }
#pragma unroll
    for (int j = 0; j < 4; ++j)
#pragma unroll
        for (int r = 0; r < 8; ++r)
            sHid[wv][r + hi][j * 16 + ch] = (_Float16)acc[j][r];
    __syncthreads();

    // layer 2: (16x64) @ (64x64); A frags = two ds_load_b128 each
    v16h a2a = fragA_lds16row(&sHid[wv][0][0], 72, 0,  lane);
    v16h a2b = fragA_lds16row(&sHid[wv][0][0], 72, 32, lane);
#pragma unroll
    for (int j = 0; j < 4; ++j) {
        v8f z = {};
        z = wmma_f16(a2a, fragB_pk(fb, FR_MLP_W2 + j * 2 + 0, lane), z);
        z = wmma_f16(a2b, fragB_pk(fb, FR_MLP_W2 + j * 2 + 1, lane), z);
        acc[j] = silu8(z);
    }
    __syncthreads();
#pragma unroll
    for (int j = 0; j < 4; ++j)
#pragma unroll
        for (int r = 0; r < 8; ++r)
            sHid[wv][r + hi][j * 16 + ch] = (_Float16)acc[j][r];
    __syncthreads();

    v16h a3a = fragA_lds16row(&sHid[wv][0][0], 72, 0,  lane);
    v16h a3b = fragA_lds16row(&sHid[wv][0][0], 72, 32, lane);

    // layer 3: (16x64) @ (64x320), consumed 16 channels at a time
    for (int j = 0; j < 4; ++j) {
        int ch0 = j * 16;
        v8f wacc[5];
#pragma unroll
        for (int p = 0; p < 5; ++p) {
            int t = p * 4 + j;
            v8f z = {};
            z = wmma_f16(a3a, fragB_pk(fb, FR_MLP_W3 + t * 2 + 0, lane), z);
            z = wmma_f16(a3b, fragB_pk(fb, FR_MLP_W3 + t * 2 + 1, lane), z);
            wacc[p] = z;
        }
        int c = ch0 + ch;
#pragma unroll
        for (int r = 0; r < 8; ++r) {
            int li = r + hi;
            int sn = sSnd[wv][li];
            int rc = sRcv[wv][li];
            float y0  = sY[wv][li][0];
            float y1x = sY[wv][li][1];
            float y1y = sY[wv][li][2];
            float y1z = sY[wv][li][3];
            const float* hp = H + (long)sn * ROW;
            float n0 = hp[c];
            float nx = hp[64 + 3 * c + 0];
            float ny = hp[64 + 3 * c + 1];
            float nz = hp[64 + 3 * c + 2];
            float w0 = wacc[0][r], w1 = wacc[1][r], w2 = wacc[2][r];
            float w3 = wacc[3][r], w4 = wacc[4][r];
            float m0 = w0 * n0 * y0 + w1 * INV3 * (nx * y1x + ny * y1y + nz * y1z);
            float cx = ny * y1z - nz * y1y;
            float cy = nz * y1x - nx * y1z;
            float cz = nx * y1y - ny * y1x;
            float m1x = w2 * n0 * y1x + w3 * nx * y0 + w4 * INV2 * cx;
            float m1y = w2 * n0 * y1y + w3 * ny * y0 + w4 * INV2 * cy;
            float m1z = w2 * n0 * y1z + w3 * nz * y0 + w4 * INV2 * cz;
            float* tp = T + (long)rc * ROW;
            atomicAdd(tp + c, m0);
            atomicAdd(tp + 64 + 3 * c + 0, m1x);
            atomicAdd(tp + 64 + 3 * c + 1, m1y);
            atomicAdd(tp + 64 + 3 * c + 2, m1z);
        }
    }
}

// lin / attr-contraction / self-connection / combine / fin
__global__ void post_kernel(const float* __restrict__ nf,   const float* __restrict__ attr,
                            const float* __restrict__ T,    const _Float16* __restrict__ fb,
                            float* __restrict__ out, int N_) {
    __shared__ __align__(16) _Float16 sOut[NW][16][264];  // 528B stride (16B aligned)
    int lane = threadIdx.x & 31;
    int wv   = threadIdx.x >> 5;
    int nBase = (blockIdx.x * NW + wv) * 16;
    if (nBase + 16 > N_) return;                  // wave-uniform exit (16 | N)
    int ch = lane & 15;
    int hi = (lane >> 4) << 3;

    // A fragments (converted to f16 once)
    v16h aT0[2], aT1[3][2], xf0[2], xf1[3][2];
    aT0[0] = fragA_rows(T, ROW, nBase, 0,  64, lane);
    aT0[1] = fragA_rows(T, ROW, nBase, 32, 64, lane);
    xf0[0] = fragA_rows(nf, ROW, nBase, 0,  64, lane);
    xf0[1] = fragA_rows(nf, ROW, nBase, 32, 64, lane);
#pragma unroll
    for (int m = 0; m < 3; ++m) {
        aT1[m][0] = fragA_vec3(T,  nBase, m, 0,  lane);
        aT1[m][1] = fragA_vec3(T,  nBase, m, 32, lane);
        xf1[m][0] = fragA_vec3(nf, nBase, m, 0,  lane);
        xf1[m][1] = fragA_vec3(nf, nBase, m, 32, lane);
    }
    v16h aAt = fragA_rows(attr, 10, nBase, 0, 10, lane);
    _Float16 ah[10];
    {
        const float* ap = attr + (long)(nBase + (lane & 15)) * 10;
#pragma unroll
        for (int a = 0; a < 10; ++a) ah[a] = (_Float16)ap[a];
    }

    for (int j = 0; j < 4; ++j) {
        int ch0 = j * 16;
        v8f u0 = {};
        u0 = wmma_f16(aT0[0], fragB_pk(fb, FR_LIN0 + j * 2 + 0, lane), u0);
        u0 = wmma_f16(aT0[1], fragB_pk(fb, FR_LIN0 + j * 2 + 1, lane), u0);
        v8f u1[3];
#pragma unroll
        for (int m = 0; m < 3; ++m) {
            v8f z = {};
            z = wmma_f16(aT1[m][0], fragB_pk(fb, FR_LIN1 + j * 2 + 0, lane), z);
            z = wmma_f16(aT1[m][1], fragB_pk(fb, FR_LIN1 + j * 2 + 1, lane), z);
            u1[m] = z;
        }
        v8f cw0[3], cw1[2];
#pragma unroll
        for (int p = 0; p < 3; ++p) {
            v8f z = {};
            cw0[p] = wmma_f16(aAt, fragB_pk(fb, FR_C0W + p * 4 + j, lane), z);
        }
#pragma unroll
        for (int p = 0; p < 2; ++p) {
            v8f z = {};
            cw1[p] = wmma_f16(aAt, fragB_pk(fb, FR_C1W + p * 4 + j, lane), z);
        }
        // self-connection: (attr ⊗ x) @ scW, K = 640 = 20 steps; A built with pk f16 muls
        v8f sc0 = {};
#pragma unroll
        for (int ks = 0; ks < 20; ++ks)
            sc0 = wmma_f16(vsplat_mul(xf0[ks & 1], ah[ks >> 1]),
                           fragB_pk(fb, FR_SCW0 + ks * 4 + j, lane), sc0);
        v8f sc1[3];
#pragma unroll
        for (int m = 0; m < 3; ++m) {
            v8f z = {};
#pragma unroll
            for (int ks = 0; ks < 20; ++ks)
                z = wmma_f16(vsplat_mul(xf1[m][ks & 1], ah[ks >> 1]),
                             fragB_pk(fb, FR_SCW1 + ks * 4 + j, lane), z);
            sc1[m] = z;
        }
        // combine (all operands share the same D-fragment layout)
        v8f nrm = u1[0] * u1[0] + u1[1] * u1[1] + u1[2] * u1[2];
        v8f o0 = cw0[0] * u0 + cw0[1] * (u0 * u0) + cw0[2] * nrm + sc0;
        _Float16* sp0 = &sOut[wv][hi][0] + ch0 + ch;
        _Float16* sp1 = &sOut[wv][hi][0] + 64 + 3 * (ch0 + ch);
#pragma unroll
        for (int r = 0; r < 8; ++r) {
            sp0[r * 264] = (_Float16)o0[r];
#pragma unroll
            for (int m = 0; m < 3; ++m) {
                float o1 = cw1[0][r] * u1[m][r] + cw1[1][r] * u0[r] * u1[m][r] + sc1[m][r];
                sp1[r * 264 + m] = (_Float16)o1;
            }
        }
    }
    __syncthreads();

    v16h aO0[2], aO1[3][2];
    aO0[0] = fragA_lds16row(&sOut[wv][0][0], 264, 0,  lane);
    aO0[1] = fragA_lds16row(&sOut[wv][0][0], 264, 32, lane);
#pragma unroll
    for (int m = 0; m < 3; ++m) {
        aO1[m][0] = fragA_lds16v(&sOut[wv][0][0], 264, 0,  lane, m);
        aO1[m][1] = fragA_lds16v(&sOut[wv][0][0], 264, 32, lane, m);
    }
    float* o0p = out + (long)(nBase + hi) * ROW + ch;           // + r*ROW + j*16
    float* o1p = out + (long)(nBase + hi) * ROW + 64 + 3 * ch;  // + r*ROW + 48*j + m
    for (int j = 0; j < 4; ++j) {
        v8f f0 = {};
        f0 = wmma_f16(aO0[0], fragB_pk(fb, FR_FIN0 + j * 2 + 0, lane), f0);
        f0 = wmma_f16(aO0[1], fragB_pk(fb, FR_FIN0 + j * 2 + 1, lane), f0);
#pragma unroll
        for (int r = 0; r < 8; ++r)
            o0p[r * ROW + j * 16] = f0[r];
#pragma unroll
        for (int m = 0; m < 3; ++m) {
            v8f f1 = {};
            f1 = wmma_f16(aO1[m][0], fragB_pk(fb, FR_FIN1 + j * 2 + 0, lane), f1);
            f1 = wmma_f16(aO1[m][1], fragB_pk(fb, FR_FIN1 + j * 2 + 1, lane), f1);
#pragma unroll
            for (int r = 0; r < 8; ++r)
                o1p[r * ROW + 48 * j + m] = f1[r];
        }
    }
}

// ---------------- launch ----------------

extern "C" void kernel_launch(void* const* d_in, const int* in_sizes, int n_in,
                              void* d_out, int out_size, void* d_ws, size_t ws_size,
                              hipStream_t stream) {
    const float* nf    = (const float*)d_in[0];
    const float* attr  = (const float*)d_in[1];
    const float* sph   = (const float*)d_in[2];
    const float* rb    = (const float*)d_in[3];
    const int*   ei    = (const int*)  d_in[4];
    const float* preW0 = (const float*)d_in[5];
    const float* preW1 = (const float*)d_in[6];
    const float* W1    = (const float*)d_in[7];
    const float* W2    = (const float*)d_in[8];
    const float* W3    = (const float*)d_in[9];
    const float* lin0  = (const float*)d_in[10];
    const float* lin1  = (const float*)d_in[11];
    const float* c0w   = (const float*)d_in[12];
    const float* c1w   = (const float*)d_in[13];
    const float* scW0  = (const float*)d_in[14];
    const float* scW1  = (const float*)d_in[15];
    const float* fin0  = (const float*)d_in[16];
    const float* fin1  = (const float*)d_in[17];
    float* outp = (float*)d_out;

    int N_ = in_sizes[0] / ROW;      // (N, 4C)
    int E_ = in_sizes[2] / 4;        // (E, 4)

    float*    H  = (float*)d_ws;                       // N*256 f32
    float*    T  = H + (size_t)N_ * ROW;               // N*256 f32 (accumulator)
    _Float16* fb = (_Float16*)(T + (size_t)N_ * ROW);  // FR_TOTAL * 512 halves

    pack_kernel<<<FR_TOTAL, 32, 0, stream>>>(preW0, preW1, W1, W2, W3, lin0, lin1,
                                             c0w, c1w, scW0, scW1, fin0, fin1, fb);

    long t4 = (long)N_ * ROW / 4;
    zero_kernel<<<(int)((t4 + 255) / 256), 256, 0, stream>>>((float4*)T, t4);

    int nBlocks = (N_ + NW * 16 - 1) / (NW * 16);
    pre_kernel<<<nBlocks, NW * 32, 0, stream>>>(nf, fb, H, N_);

    int eBlocks = (E_ + NW * 16 - 1) / (NW * 16);
    edge_kernel<<<eBlocks, NW * 32, 0, stream>>>(H, sph, rb, ei, fb, T, E_);

    post_kernel<<<nBlocks, NW * 32, 0, stream>>>(nf, attr, T, fb, outp, N_);
}
